// MinGRU_13761075216774
// MI455X (gfx1250) — matrix operand order repaired and checked by reading, not compiled, not run on hardware
//
#include <hip/hip_runtime.h>
#include <hip/hip_bf16.h>
#include <math.h>

// ---------------- problem constants ----------------
#define B_DIM 8
#define T_DIM 2048
#define D_DIM 2048
#define M_DIM (B_DIM * T_DIM)     // 16384 rows of x
#define CCH   16                  // scan chunks along T
#define LCH   (T_DIM / CCH)       // 128 steps per chunk

// ---------------- vector types ----------------
typedef __attribute__((ext_vector_type(16))) __bf16 v16bf;
typedef __attribute__((ext_vector_type(8)))  __bf16 v8bf;
typedef __attribute__((ext_vector_type(8)))  float  v8f;
typedef __attribute__((ext_vector_type(8)))  unsigned short v8u;

#if __has_builtin(__builtin_amdgcn_global_load_async_to_lds_b128)
#define HAVE_ASYNC_LDS 1
#endif

// builtin wants: (global int4*, shared int4*, imm offset, imm cpol)
typedef int v4i_gcc __attribute__((vector_size(16)));
typedef __attribute__((address_space(1))) v4i_gcc* gv4_t;
typedef __attribute__((address_space(3))) v4i_gcc* lv4_t;

// async 16B global -> LDS copy (ASYNCcnt) with safe fallback
__device__ __forceinline__ void copy16_g2l(const unsigned short* g, unsigned short* l) {
#ifdef HAVE_ASYNC_LDS
    __builtin_amdgcn_global_load_async_to_lds_b128(
        (gv4_t)(uintptr_t)g,
        (lv4_t)(unsigned)(uintptr_t)l,   // low 32 bits of generic LDS addr == LDS offset
        0, 0);
#else
    *reinterpret_cast<v8u*>(l) = *reinterpret_cast<const v8u*>(g);
#endif
}

__device__ __forceinline__ void wait_async_done() {
#ifdef HAVE_ASYNC_LDS
#if __has_builtin(__builtin_amdgcn_s_wait_asynccnt)
    __builtin_amdgcn_s_wait_asynccnt(0);
#else
    asm volatile("s_wait_asynccnt 0x0" ::: "memory");
#endif
#endif
}

// fp32 -> bf16 round-to-nearest-even
__device__ __forceinline__ unsigned short f2bf(float f) {
    unsigned int u = __float_as_uint(f);
    unsigned int r = u + 0x7fffu + ((u >> 16) & 1u);
    return (unsigned short)(r >> 16);
}

// ---------------- kernel 1: fp32 -> bf16 conversion ----------------
__global__ __launch_bounds__(256) void mingru_cvt_bf16(
    const float* __restrict__ src, unsigned short* __restrict__ dst, int n)
{
    int i = (blockIdx.x * 256 + threadIdx.x) * 8;
    if (i >= n) return;
    float4 v0 = *(const float4*)(src + i);
    float4 v1 = *(const float4*)(src + i + 4);
    v8u o;
    o[0] = f2bf(v0.x); o[1] = f2bf(v0.y); o[2] = f2bf(v0.z); o[3] = f2bf(v0.w);
    o[4] = f2bf(v1.x); o[5] = f2bf(v1.y); o[6] = f2bf(v1.z); o[7] = f2bf(v1.w);
    *(v8u*)(dst + i) = o;
}

// ---------------- kernel 2: dual GEMM + gate epilogue ----------------
// C_z[m,e] = sum_d x[m,d]*Wz[e,d];  C_h[m,e] = sum_d x[m,d]*Wh[e,d]
// a_buf = 1 - sigmoid(C_z + bz);  b_buf = sigmoid(C_z + bz) * (C_h + bh)
__global__ __launch_bounds__(256) void mingru_gemm_wmma(
    const unsigned short* __restrict__ xb,
    const unsigned short* __restrict__ wzb,
    const unsigned short* __restrict__ whb,
    const float* __restrict__ bz,
    const float* __restrict__ bh,
    float* __restrict__ a_buf,
    float* __restrict__ bb_buf)
{
    // double-buffered LDS tiles, rows padded 64->72 bf16 to stagger banks
    __shared__ __attribute__((aligned(16))) unsigned short lA [2][128][72];
    __shared__ __attribute__((aligned(16))) unsigned short lBz[2][ 64][72];
    __shared__ __attribute__((aligned(16))) unsigned short lBh[2][ 64][72];

    const int tid = threadIdx.x;
    const int n0  = blockIdx.x * 64;    // output column tile (e)
    const int m0  = blockIdx.y * 128;   // output row tile (b*T+t)

    const int lane = tid & 31;
    const int wave = tid >> 5;          // 0..7
    const int wm   = wave >> 1;         // 0..3 : 32-row strip
    const int wn   = wave & 1;          // 0..1 : 32-col strip
    const int l15  = lane & 15;
    const int lh   = lane >> 4;         // lane half

    v8f accz[2][2], acch[2][2];
#pragma unroll
    for (int mi = 0; mi < 2; ++mi)
#pragma unroll
        for (int ni = 0; ni < 2; ++ni) {
            accz[mi][ni] = (v8f)0.0f;
            acch[mi][ni] = (v8f)0.0f;
        }

    // ---- stage one K-slab (64 wide) into LDS buffer p ----
    auto issue_tile = [&](int k0, int p) {
#ifdef HAVE_ASYNC_LDS
#pragma unroll
        for (int i = 0; i < 4; ++i) {                 // A: 128x64, 4 chunks/thread
            int ch = tid + i * 256;
            int r = ch >> 3, c8 = (ch & 7) * 8;
            copy16_g2l(xb + (size_t)(m0 + r) * D_DIM + k0 + c8, &lA[p][r][c8]);
        }
#pragma unroll
        for (int i = 0; i < 2; ++i) {                 // Bz/Bh: 64x64, 2 chunks/thread
            int ch = tid + i * 256;
            int r = ch >> 3, c8 = (ch & 7) * 8;
            copy16_g2l(wzb + (size_t)(n0 + r) * D_DIM + k0 + c8, &lBz[p][r][c8]);
            copy16_g2l(whb + (size_t)(n0 + r) * D_DIM + k0 + c8, &lBh[p][r][c8]);
        }
#else
        // batched fallback: all global loads in flight before any LDS store
        v8u ra[4], rz[2], rh[2];
#pragma unroll
        for (int i = 0; i < 4; ++i) {
            int ch = tid + i * 256;
            int r = ch >> 3, c8 = (ch & 7) * 8;
            ra[i] = *reinterpret_cast<const v8u*>(xb + (size_t)(m0 + r) * D_DIM + k0 + c8);
        }
#pragma unroll
        for (int i = 0; i < 2; ++i) {
            int ch = tid + i * 256;
            int r = ch >> 3, c8 = (ch & 7) * 8;
            rz[i] = *reinterpret_cast<const v8u*>(wzb + (size_t)(n0 + r) * D_DIM + k0 + c8);
            rh[i] = *reinterpret_cast<const v8u*>(whb + (size_t)(n0 + r) * D_DIM + k0 + c8);
        }
#pragma unroll
        for (int i = 0; i < 4; ++i) {
            int ch = tid + i * 256;
            int r = ch >> 3, c8 = (ch & 7) * 8;
            *reinterpret_cast<v8u*>(&lA[p][r][c8]) = ra[i];
        }
#pragma unroll
        for (int i = 0; i < 2; ++i) {
            int ch = tid + i * 256;
            int r = ch >> 3, c8 = (ch & 7) * 8;
            *reinterpret_cast<v8u*>(&lBz[p][r][c8]) = rz[i];
            *reinterpret_cast<v8u*>(&lBh[p][r][c8]) = rh[i];
        }
#endif
    };

    // ---- run 2 WMMA K-steps against LDS buffer p ----
    auto compute_tile = [&](int p) {
#pragma unroll
        for (int kk = 0; kk < 64; kk += 32) {
            v16bf af[2];
#pragma unroll
            for (int mi = 0; mi < 2; ++mi) {          // A frag: 16x32, contiguous k = 16*lh + s (B same)
                int r = wm * 32 + mi * 16 + l15;
                int c = kk + lh * 16;
                v8bf lo = *reinterpret_cast<const v8bf*>(&lA[p][r][c]);
                v8bf hi = *reinterpret_cast<const v8bf*>(&lA[p][r][c + 8]);
                af[mi] = __builtin_shufflevector(lo, hi,
                         0,1,2,3,4,5,6,7,8,9,10,11,12,13,14,15);
            }
            v16bf bzf[2], bhf[2];
#pragma unroll
            for (int ni = 0; ni < 2; ++ni) {          // B frag: 32x16, B[k,n]=W[n,k]
                int r = wn * 32 + ni * 16 + l15;
                int c = kk + lh * 16;
                v8bf lo = *reinterpret_cast<const v8bf*>(&lBz[p][r][c]);
                v8bf hi = *reinterpret_cast<const v8bf*>(&lBz[p][r][c + 8]);
                bzf[ni] = __builtin_shufflevector(lo, hi,
                          0,1,2,3,4,5,6,7,8,9,10,11,12,13,14,15);
                lo = *reinterpret_cast<const v8bf*>(&lBh[p][r][c]);
                hi = *reinterpret_cast<const v8bf*>(&lBh[p][r][c + 8]);
                bhf[ni] = __builtin_shufflevector(lo, hi,
                          0,1,2,3,4,5,6,7,8,9,10,11,12,13,14,15);
            }
#pragma unroll
            for (int mi = 0; mi < 2; ++mi)
#pragma unroll
                for (int ni = 0; ni < 2; ++ni) {
                    accz[mi][ni] = __builtin_amdgcn_wmma_f32_16x16x32_bf16(
                        false, af[mi], false, bzf[ni], (short)0, accz[mi][ni], false, false);
                    acch[mi][ni] = __builtin_amdgcn_wmma_f32_16x16x32_bf16(
                        false, af[mi], false, bhf[ni], (short)0, acch[mi][ni], false, false);
                }
        }
    };

    // ---- software pipeline over K, unrolled by 2 slabs (ping-pong constant) ----
    issue_tile(0, 0);
    for (int k0 = 0; k0 < D_DIM; k0 += 128) {
        // phase 0: consume buf 0 (slab k0), prefetch slab k0+64 into buf 1
        wait_async_done();
        __syncthreads();
        if (k0 + 64 < D_DIM) issue_tile(k0 + 64, 1);
        compute_tile(0);

        // phase 1: consume buf 1 (slab k0+64), prefetch slab k0+128 into buf 0
        if (k0 + 64 < D_DIM) {
            wait_async_done();
            __syncthreads();
            if (k0 + 128 < D_DIM) issue_tile(k0 + 128, 0);
            compute_tile(1);
        }
    }

    // ---- epilogue: bias, sigmoid gate, write a = 1-z and b = z*h~ ----
#pragma unroll
    for (int ni = 0; ni < 2; ++ni) {
        int e = n0 + wn * 32 + ni * 16 + l15;     // C/D layout: lane -> column
        float bzv = bz[e], bhv = bh[e];
#pragma unroll
        for (int mi = 0; mi < 2; ++mi) {
            int mbase = m0 + wm * 32 + mi * 16 + lh * 8;  // vgpr r -> row (+8 upper half)
#pragma unroll
            for (int r = 0; r < 8; ++r) {
                int m = mbase + r;
                float zval = accz[mi][ni][r] + bzv;
                zval = 1.0f / (1.0f + __expf(-zval));
                float ht = acch[mi][ni][r] + bhv;
                size_t idx = (size_t)m * D_DIM + e;
                a_buf[idx]  = 1.0f - zval;
                bb_buf[idx] = zval * ht;
            }
        }
    }
}

// ---------------- kernel 3: per-chunk affine aggregate ----------------
__global__ __launch_bounds__(256) void mingru_scan_chunks(
    const float* __restrict__ a_buf, const float* __restrict__ b_buf,
    float* __restrict__ Aagg, float* __restrict__ Bagg)
{
    int id = blockIdx.x * 256 + threadIdx.x;       // (b*CCH + c)*D + d
    int d  = id % D_DIM;
    int bc = id / D_DIM;
    int c  = bc % CCH;
    int b  = bc / CCH;
    size_t base = ((size_t)b * T_DIM + (size_t)c * LCH) * D_DIM + d;
    float A = 1.0f, Bv = 0.0f;
#pragma unroll 8
    for (int t = 0; t < LCH; ++t) {
        float a  = __builtin_nontemporal_load(a_buf + base + (size_t)t * D_DIM);
        float bb = __builtin_nontemporal_load(b_buf + base + (size_t)t * D_DIM);
        A  = a * A;
        Bv = fmaf(a, Bv, bb);
    }
    Aagg[id] = A;
    Bagg[id] = Bv;
}

// ---------------- kernel 4: combine chunk aggregates ----------------
__global__ __launch_bounds__(256) void mingru_scan_combine(
    const float* __restrict__ Aagg, const float* __restrict__ Bagg,
    const float* __restrict__ h_prev,
    float* __restrict__ Hstart, float* __restrict__ last)
{
    int id = blockIdx.x * 256 + threadIdx.x;       // b*D + d
    int d  = id % D_DIM;
    int b  = id / D_DIM;
    float h = h_prev[id];
#pragma unroll
    for (int c = 0; c < CCH; ++c) {
        int aid = (b * CCH + c) * D_DIM + d;
        Hstart[aid] = h;
        h = fmaf(Aagg[aid], h, Bagg[aid]);
    }
    last[id] = h;   // out[:, -1, :]
}

// ---------------- kernel 5: emit outputs within each chunk ----------------
__global__ __launch_bounds__(256) void mingru_scan_emit(
    const float* __restrict__ a_buf, const float* __restrict__ b_buf,
    const float* __restrict__ Hstart, float* __restrict__ out)
{
    int id = blockIdx.x * 256 + threadIdx.x;       // (b*CCH + c)*D + d
    int d  = id % D_DIM;
    int bc = id / D_DIM;
    int c  = bc % CCH;
    int b  = bc / CCH;
    size_t base = ((size_t)b * T_DIM + (size_t)c * LCH) * D_DIM + d;
    float h = Hstart[id];
#pragma unroll 8
    for (int t = 0; t < LCH; ++t) {
        float a  = __builtin_nontemporal_load(a_buf + base + (size_t)t * D_DIM);
        float bb = __builtin_nontemporal_load(b_buf + base + (size_t)t * D_DIM);
        h = fmaf(a, h, bb);
        __builtin_nontemporal_store(h, out + base + (size_t)t * D_DIM);
    }
}

// ---------------- launcher ----------------
extern "C" void kernel_launch(void* const* d_in, const int* in_sizes, int n_in,
                              void* d_out, int out_size, void* d_ws, size_t ws_size,
                              hipStream_t stream)
{
    (void)in_sizes; (void)n_in; (void)out_size; (void)ws_size;
    const float* x      = (const float*)d_in[0];
    const float* h_prev = (const float*)d_in[1];
    const float* Wz     = (const float*)d_in[2];
    const float* bz     = (const float*)d_in[3];
    const float* Wh     = (const float*)d_in[4];
    const float* bh     = (const float*)d_in[5];

    char* ws = (char*)d_ws;
    unsigned short* xb   = (unsigned short*)(ws);                    // 64 MiB
    unsigned short* wzb  = (unsigned short*)(ws + 67108864ULL);      //  8 MiB
    unsigned short* whb  = (unsigned short*)(ws + 75497472ULL);      //  8 MiB
    float* a_buf = (float*)(ws + 83886080ULL);                       // 128 MiB
    float* b_buf = (float*)(ws + 218103808ULL);                      // 128 MiB
    float* Aagg  = (float*)(ws + 352321536ULL);                      //  1 MiB
    float* Bagg  = (float*)(ws + 353370112ULL);                      //  1 MiB
    float* Hst   = (float*)(ws + 354418688ULL);                      //  1 MiB

    float* out  = (float*)d_out;
    float* last = out + (size_t)M_DIM * D_DIM;

    // 1) fp32 -> bf16
    mingru_cvt_bf16<<<(M_DIM * D_DIM) / 8 / 256, 256, 0, stream>>>(x,  xb,  M_DIM * D_DIM);
    mingru_cvt_bf16<<<(D_DIM * D_DIM) / 8 / 256, 256, 0, stream>>>(Wz, wzb, D_DIM * D_DIM);
    mingru_cvt_bf16<<<(D_DIM * D_DIM) / 8 / 256, 256, 0, stream>>>(Wh, whb, D_DIM * D_DIM);

    // 2) dual WMMA GEMM + gate epilogue
    dim3 ggrid(D_DIM / 64, M_DIM / 128);
    mingru_gemm_wmma<<<ggrid, 256, 0, stream>>>(xb, wzb, whb, bz, bh, a_buf, b_buf);

    // 3-5) chunked parallel scan
    mingru_scan_chunks <<<(B_DIM * CCH * D_DIM) / 256, 256, 0, stream>>>(a_buf, b_buf, Aagg, Bagg);
    mingru_scan_combine<<<(B_DIM * D_DIM) / 256,        256, 0, stream>>>(Aagg, Bagg, h_prev, Hst, last);
    mingru_scan_emit   <<<(B_DIM * CCH * D_DIM) / 256,  256, 0, stream>>>(a_buf, b_buf, Hst, out);
}
